// MixExpertAttentionSACNetwork_6579889897872
// MI455X (gfx1250) — matrix operand order repaired and checked by reading, not compiled, not run on hardware
//
#include <hip/hip_runtime.h>
#include <hip/hip_bf16.h>

typedef __bf16 bf16_t;
typedef __attribute__((ext_vector_type(16))) __bf16 v16bf;
typedef __attribute__((ext_vector_type(8)))  float  v8f;

// Problem constants
#define BATCH 4096
#define OBS_D 390
#define ACT_D 20
#define XIN_D 410     // OBS+ACT
#define XPAD_D 416    // padded to multiple of 32 (zero-filled tail)
#define H1_D 1024
#define H2_D 512
#define EMB_D 256
#define NEXP 8

// GEMM tiling: 256 threads = 8 waves; block tile 128x64, K step 32.
// Waves: 4 along M (32 rows each), 2 along N (32 cols each); each wave does 2x2 WMMA 16x16 tiles.
#define BM 128
#define BN 64
#define BK 32
#define LDA_S 40   // padded LDS row (bf16 elems) for A tile [BM][LDA_S]
#define LDB_S 40   // padded LDS row for B^T tile [BN][LDB_S]

// ---------------------------------------------------------------------------
// Elementwise fp32 -> bf16 converter (grid-stride)
// ---------------------------------------------------------------------------
__global__ void f32_to_bf16_kernel(const float* __restrict__ src,
                                   bf16_t* __restrict__ dst, long long n) {
  long long i = (long long)blockIdx.x * blockDim.x + threadIdx.x;
  long long stride = (long long)gridDim.x * blockDim.x;
  for (; i < n; i += stride) dst[i] = (bf16_t)src[i];
}

// Concatenate [B,390] and [B,20] into bf16 [B,416] (cols 410..415 zero)
__global__ void concat_bf16_kernel(const float* __restrict__ sf,
                                   const float* __restrict__ ac,
                                   bf16_t* __restrict__ X) {
  long long n = (long long)BATCH * XPAD_D;
  long long i = (long long)blockIdx.x * blockDim.x + threadIdx.x;
  long long stride = (long long)gridDim.x * blockDim.x;
  for (; i < n; i += stride) {
    int b = (int)(i / XPAD_D);
    int c = (int)(i % XPAD_D);
    float v;
    if (c < OBS_D)      v = sf[(long long)b * OBS_D + c];
    else if (c < XIN_D) v = ac[(long long)b * ACT_D + (c - OBS_D)];
    else                v = 0.0f;
    X[i] = (bf16_t)v;
  }
}

// rep_W1 [410,1024] fp32 -> [416,1024] bf16 with zero rows 410..415
__global__ void w1_pad_bf16_kernel(const float* __restrict__ w,
                                   bf16_t* __restrict__ dst) {
  long long n = (long long)XPAD_D * H1_D;
  long long i = (long long)blockIdx.x * blockDim.x + threadIdx.x;
  long long stride = (long long)gridDim.x * blockDim.x;
  for (; i < n; i += stride) {
    int r = (int)(i / H1_D);
    dst[i] = (r < XIN_D) ? (bf16_t)w[i] : (bf16_t)0.0f;
  }
}

// ---------------------------------------------------------------------------
// Tiled bf16 WMMA GEMM:  O[z] = act( A[z][M,K] * W[z][K,N] + bias[z][N] )
// Requires: M % 128 == 0, N % 64 == 0, K % 32 == 0 (no bounds checks).
// ---------------------------------------------------------------------------
template <int RELU, int OUT_BF16>
__global__ __launch_bounds__(256)
void gemm_bf16_wmma(const bf16_t* __restrict__ Ag, const bf16_t* __restrict__ Wg,
                    const float* __restrict__ biasg, void* __restrict__ Og,
                    int M, int N, int K,
                    long long aZ, long long wZ, long long bZ, long long oZ) {
  __shared__ __align__(16) bf16_t As[BM * LDA_S];
  __shared__ __align__(16) bf16_t Bs[BN * LDB_S];

  const int z = blockIdx.z;
  const bf16_t* A = Ag + z * aZ;
  const bf16_t* W = Wg + z * wZ;
  const float* bias = biasg + z * bZ;

  const int tid = threadIdx.x;
  const int lane = tid & 31;
  const int wave = tid >> 5;
  const int wm = wave & 3;   // wave M position (0..3) -> 32 rows each
  const int wn = wave >> 2;  // wave N position (0..1) -> 32 cols each
  const int m0 = blockIdx.y * BM;
  const int n0 = blockIdx.x * BN;

  v8f acc[2][2] = {};

  // A-tile load map: 128x32 over 256 threads => one 4xbf16 (8B) vector per row-step
  const int ar = tid >> 3;        // base row 0..31 (+32 steps)
  const int ac4 = (tid & 7) * 4;  // 4-col group (8B aligned since K%4==0)
  // B-tile load map: 32x64 over 256 threads => one 8xbf16 (16B) vector each
  const int bkr = tid >> 3;       // K row 0..31
  const int bnc = (tid & 7) * 8;  // 8-col group (16B aligned since N%8==0)

  const int nk = K / BK;
  for (int kt = 0; kt < nk; ++kt) {
    const int k0 = kt * BK;
    // ---- stage A tile to LDS (row-major), wide loads/stores ----
#pragma unroll
    for (int rr = 0; rr < 4; ++rr) {
      const int r = ar + rr * 32;
      const uint2 v = *(const uint2*)(A + (long long)(m0 + r) * K + k0 + ac4);
      *(uint2*)(&As[r * LDA_S + ac4]) = v;
    }
    // ---- stage B tile to LDS transposed: Bs[n][k] ----
    {
      const int k = k0 + bkr;
      union { uint4 u; bf16_t h[8]; } t;
      t.u = *(const uint4*)(W + (long long)k * N + n0 + bnc);
#pragma unroll
      for (int cc = 0; cc < 8; ++cc)
        Bs[(bnc + cc) * LDB_S + bkr] = t.h[cc];
    }
    __syncthreads();

    // ---- build fragments per documented 16-bit WMMA layouts ----
    // A 16x32: lanes 0-15 hold row M=lane, elems 0..7 = K0..7, 8..15 = K16..23;
    //          lanes 16-31 hold row M=lane-16, elems 0..7 = K8..15, 8..15 = K24..31.
    v16bf afrag[2], bfrag[2];
    const int ks = (lane >> 4) * 8;
#pragma unroll
    for (int i = 0; i < 2; ++i) {
      const int row = wm * 32 + i * 16 + (lane & 15);
      const bf16_t* p0 = &As[row * LDA_S + ks];
      const bf16_t* p1 = &As[row * LDA_S + 16 + ks];
#pragma unroll
      for (int e = 0; e < 8; ++e) { afrag[i][e] = p0[e]; afrag[i][8 + e] = p1[e]; }
    }
    // B 32x16: lanes 0-15 hold col N=lane, elems 0..7 = K0..7, 8..15 = K16..23;
    //          lanes 16-31 hold col N=lane-16, elems 0..7 = K8..15, 8..15 = K24..31.
    const int kbs = (lane >> 4) * 8;
#pragma unroll
    for (int j = 0; j < 2; ++j) {
      const int col = wn * 32 + j * 16 + (lane & 15);
      const bf16_t* p = &Bs[col * LDB_S + kbs];
#pragma unroll
      for (int e = 0; e < 8; ++e) { bfrag[j][e] = p[e]; bfrag[j][8 + e] = p[16 + e]; }
    }

#pragma unroll
    for (int i = 0; i < 2; ++i)
#pragma unroll
      for (int j = 0; j < 2; ++j)
        acc[i][j] = __builtin_amdgcn_wmma_f32_16x16x32_bf16(
            false, afrag[i], false, bfrag[j], (short)0, acc[i][j], false, false);
    __syncthreads();
  }

  // ---- epilogue: bias (+ReLU), store. C/D layout: elem e -> M = base + e + 8*(lane>=16),
  //      N = base + (lane&15). ----
#pragma unroll
  for (int i = 0; i < 2; ++i) {
#pragma unroll
    for (int j = 0; j < 2; ++j) {
      const int n = n0 + wn * 32 + j * 16 + (lane & 15);
      const float bv = bias[n];
      const int mbase = m0 + wm * 32 + i * 16 + (lane >> 4) * 8;
#pragma unroll
      for (int e = 0; e < 8; ++e) {
        float v = acc[i][j][e] + bv;
        if (RELU) v = fmaxf(v, 0.0f);
        const long long idx = (long long)(mbase + e) * N + n;
        if (OUT_BF16)
          ((bf16_t*)Og)[z * oZ + idx] = (bf16_t)v;
        else
          ((float*)Og)[z * oZ + idx] = v;
      }
    }
  }
}

// ---------------------------------------------------------------------------
// Attention: query = tanh(emb[task]); scores over E=8; softmax; value mix;
// expert loss accumulation. One 256-thread block per batch row.
// ---------------------------------------------------------------------------
__global__ __launch_bounds__(256)
void attention_kernel(const int* __restrict__ task_id, const float* __restrict__ emb,
                      const float* __restrict__ keys,  // [E,B,D] fp32
                      const float* __restrict__ vals,  // [E,B,D] fp32
                      bf16_t* __restrict__ tin,        // [B,D] bf16
                      float* __restrict__ loss_acc) {
  __shared__ float qs[EMB_D];
  __shared__ float sc[NEXP];
  const int b = blockIdx.x;
  const int d = threadIdx.x;
  const int t = task_id[b];
  const float q = tanhf(emb[(long long)t * EMB_D + d]);
  qs[d] = q;
  __syncthreads();

  const int e = d >> 5, lane = d & 31;  // 8 waves -> 8 experts
  float s = 0.0f;
  const float* kp = keys + ((long long)e * BATCH + b) * EMB_D;
#pragma unroll
  for (int i = lane; i < EMB_D; i += 32) s += qs[i] * kp[i];
#pragma unroll
  for (int off = 16; off; off >>= 1) s += __shfl_xor(s, off, 32);
  if (lane == 0) sc[e] = s;
  __syncthreads();

  float mx = sc[0];
#pragma unroll
  for (int i = 1; i < NEXP; ++i) mx = fmaxf(mx, sc[i]);
  float w[NEXP], den = 0.0f;
#pragma unroll
  for (int i = 0; i < NEXP; ++i) { w[i] = expf(sc[i] - mx); den += w[i]; }
  const float inv = 1.0f / den;

  float accum = 0.0f;
#pragma unroll
  for (int i = 0; i < NEXP; ++i)
    accum += (w[i] * inv) * vals[((long long)i * BATCH + b) * EMB_D + d];
  tin[(long long)b * EMB_D + d] = (bf16_t)accum;

  if (d == 0) {
    float ls = 0.0f;
#pragma unroll
    for (int i = 0; i < NEXP; ++i) {
      float lw = logf(w[i] * inv + 1e-10f);
      ls += fminf(fmaxf(lw, -6.0f), 0.0f);
    }
    atomicAdd(loss_acc, ls);
  }
}

// Final N=1 layer: q[b] = dot(T2[b,:], tW3) + tb3 ; one wave per row.
__global__ __launch_bounds__(256)
void tower_out_kernel(const bf16_t* __restrict__ T2, const float* __restrict__ tW3,
                      const float* __restrict__ tb3, float* __restrict__ out) {
  const int gid = blockIdx.x * blockDim.x + threadIdx.x;
  const int b = gid >> 5;
  const int lane = gid & 31;
  if (b >= BATCH) return;
  float s = 0.0f;
#pragma unroll
  for (int i = lane; i < EMB_D; i += 32)
    s += (float)T2[(long long)b * EMB_D + i] * tW3[i];
#pragma unroll
  for (int off = 16; off; off >>= 1) s += __shfl_xor(s, off, 32);
  if (lane == 0) out[b] = s + tb3[0];
}

__global__ void loss_finalize_kernel(const float* __restrict__ acc, float* __restrict__ out) {
  out[0] = -0.3f * acc[0] / (float)BATCH;
}

// ---------------------------------------------------------------------------
extern "C" void kernel_launch(void* const* d_in, const int* in_sizes, int n_in,
                              void* d_out, int out_size, void* d_ws, size_t ws_size,
                              hipStream_t stream) {
  (void)in_sizes; (void)n_in; (void)out_size; (void)ws_size;
  const float* state_feat = (const float*)d_in[0];
  const float* act_in     = (const float*)d_in[1];
  const int*   task_id    = (const int*)d_in[2];
  const float* rep_W1 = (const float*)d_in[3];
  const float* rep_b1 = (const float*)d_in[4];
  const float* rep_W2 = (const float*)d_in[5];
  const float* rep_b2 = (const float*)d_in[6];
  const float* emb    = (const float*)d_in[7];
  const float* kW1 = (const float*)d_in[8];
  const float* kb1 = (const float*)d_in[9];
  const float* kW2 = (const float*)d_in[10];
  const float* kb2 = (const float*)d_in[11];
  const float* vW1 = (const float*)d_in[12];
  const float* vb1 = (const float*)d_in[13];
  const float* vW2 = (const float*)d_in[14];
  const float* vb2 = (const float*)d_in[15];
  const float* tW1 = (const float*)d_in[16];
  const float* tb1 = (const float*)d_in[17];
  const float* tW2 = (const float*)d_in[18];
  const float* tb2 = (const float*)d_in[19];
  const float* tW3 = (const float*)d_in[20];
  const float* tb3 = (const float*)d_in[21];

  // ---- workspace carve-out (256B aligned) ----
  char* ws = (char*)d_ws;
  size_t off = 0;
  auto carve = [&](size_t bytes) -> char* {
    char* p = ws + off;
    off = (off + bytes + 255) & ~(size_t)255;
    return p;
  };
  bf16_t* Xbf   = (bf16_t*)carve((size_t)BATCH * XPAD_D * 2);
  bf16_t* W1bf  = (bf16_t*)carve((size_t)XPAD_D * H1_D * 2);
  bf16_t* W2bf  = (bf16_t*)carve((size_t)H1_D * H1_D * 2);
  bf16_t* kW1bf = (bf16_t*)carve((size_t)NEXP * H1_D * H2_D * 2);
  bf16_t* kW2bf = (bf16_t*)carve((size_t)NEXP * H2_D * EMB_D * 2);
  bf16_t* vW1bf = (bf16_t*)carve((size_t)NEXP * H1_D * H2_D * 2);
  bf16_t* vW2bf = (bf16_t*)carve((size_t)NEXP * H2_D * EMB_D * 2);
  bf16_t* tW1bf = (bf16_t*)carve((size_t)EMB_D * EMB_D * 2);
  bf16_t* tW2bf = (bf16_t*)carve((size_t)EMB_D * EMB_D * 2);
  bf16_t* H1buf = (bf16_t*)carve((size_t)BATCH * H1_D * 2);
  bf16_t* REP   = (bf16_t*)carve((size_t)BATCH * H1_D * 2);
  bf16_t* HKV   = (bf16_t*)carve((size_t)NEXP * BATCH * H2_D * 2);  // reused: HK then HV
  float*  KEYS  = (float*)carve((size_t)NEXP * BATCH * EMB_D * 4);
  float*  VALS  = (float*)carve((size_t)NEXP * BATCH * EMB_D * 4);
  bf16_t* TIN   = (bf16_t*)carve((size_t)BATCH * EMB_D * 2);
  bf16_t* T1    = (bf16_t*)carve((size_t)BATCH * EMB_D * 2);
  bf16_t* T2    = (bf16_t*)carve((size_t)BATCH * EMB_D * 2);
  float*  lossA = (float*)carve(4);

  const dim3 blk(256);
  auto conv = [&](const float* s, bf16_t* d, long long n) {
    long long g = (n + 255) / 256;
    if (g > 4096) g = 4096;
    f32_to_bf16_kernel<<<dim3((unsigned)g), blk, 0, stream>>>(s, d, n);
  };

  // ---- weight / input conversions ----
  concat_bf16_kernel<<<dim3(6656), blk, 0, stream>>>(state_feat, act_in, Xbf);
  w1_pad_bf16_kernel<<<dim3(1664), blk, 0, stream>>>(rep_W1, W1bf);
  conv(rep_W2, W2bf, (long long)H1_D * H1_D);
  conv(kW1, kW1bf, (long long)NEXP * H1_D * H2_D);
  conv(kW2, kW2bf, (long long)NEXP * H2_D * EMB_D);
  conv(vW1, vW1bf, (long long)NEXP * H1_D * H2_D);
  conv(vW2, vW2bf, (long long)NEXP * H2_D * EMB_D);
  conv(tW1, tW1bf, (long long)EMB_D * EMB_D);
  conv(tW2, tW2bf, (long long)EMB_D * EMB_D);

  // ---- representation MLP ----
  gemm_bf16_wmma<1, 1><<<dim3(H1_D / BN, BATCH / BM, 1), blk, 0, stream>>>(
      Xbf, W1bf, rep_b1, H1buf, BATCH, H1_D, XPAD_D, 0, 0, 0, 0);
  gemm_bf16_wmma<0, 1><<<dim3(H1_D / BN, BATCH / BM, 1), blk, 0, stream>>>(
      H1buf, W2bf, rep_b2, REP, BATCH, H1_D, H1_D, 0, 0, 0, 0);

  // ---- expert key path (batched over z = expert) ----
  gemm_bf16_wmma<1, 1><<<dim3(H2_D / BN, BATCH / BM, NEXP), blk, 0, stream>>>(
      REP, kW1bf, kb1, HKV, BATCH, H2_D, H1_D,
      0, (long long)H1_D * H2_D, H2_D, (long long)BATCH * H2_D);
  gemm_bf16_wmma<0, 0><<<dim3(EMB_D / BN, BATCH / BM, NEXP), blk, 0, stream>>>(
      HKV, kW2bf, kb2, KEYS, BATCH, EMB_D, H2_D,
      (long long)BATCH * H2_D, (long long)H2_D * EMB_D, EMB_D, (long long)BATCH * EMB_D);

  // ---- expert value path (reuses HKV buffer) ----
  gemm_bf16_wmma<1, 1><<<dim3(H2_D / BN, BATCH / BM, NEXP), blk, 0, stream>>>(
      REP, vW1bf, vb1, HKV, BATCH, H2_D, H1_D,
      0, (long long)H1_D * H2_D, H2_D, (long long)BATCH * H2_D);
  gemm_bf16_wmma<0, 0><<<dim3(EMB_D / BN, BATCH / BM, NEXP), blk, 0, stream>>>(
      HKV, vW2bf, vb2, VALS, BATCH, EMB_D, H2_D,
      (long long)BATCH * H2_D, (long long)H2_D * EMB_D, EMB_D, (long long)BATCH * EMB_D);

  // ---- attention / softmax / value mixing / expert loss ----
  hipMemsetAsync(lossA, 0, sizeof(float), stream);
  attention_kernel<<<dim3(BATCH), blk, 0, stream>>>(task_id, emb, KEYS, VALS, TIN, lossA);

  // ---- tower ----
  gemm_bf16_wmma<1, 1><<<dim3(EMB_D / BN, BATCH / BM, 1), blk, 0, stream>>>(
      TIN, tW1bf, tb1, T1, BATCH, EMB_D, EMB_D, 0, 0, 0, 0);
  gemm_bf16_wmma<1, 1><<<dim3(EMB_D / BN, BATCH / BM, 1), blk, 0, stream>>>(
      T1, tW2bf, tb2, T2, BATCH, EMB_D, EMB_D, 0, 0, 0, 0);
  tower_out_kernel<<<dim3((BATCH * 32) / 256), blk, 0, stream>>>(
      T2, tW3, tb3, (float*)d_out);
  loss_finalize_kernel<<<dim3(1), dim3(1), 0, stream>>>(lossA, (float*)d_out + BATCH);
}
